// Attention_27977416966734
// MI455X (gfx1250) — matrix operand order repaired and checked by reading, not compiled, not run on hardware
//
#include <hip/hip_runtime.h>
#include <hip/hip_fp16.h>

typedef __attribute__((ext_vector_type(16))) _Float16 v16h;
typedef __attribute__((ext_vector_type(8)))  _Float16 v8h;
typedef __attribute__((ext_vector_type(8)))  float    v8f;

#define C_DIM 384
#define N_HEADS 6
#define DH 64
#define NB 32
#define QTOK 1025
#define QPAD 1040          // 65*16
#define KTOK 257
#define KPAD 288           // 9*32
#define ATT_SCALE 0.05103103630798287f  // 384^-0.5

// ---------------------------------------------------------------------------
// Weight transpose/pack: WT[n*384 + k] = (f16) W[k*384 + n]
// ---------------------------------------------------------------------------
__global__ void pack_wt_kernel(const float* __restrict__ W, _Float16* __restrict__ WT) {
    int idx = blockIdx.x * blockDim.x + threadIdx.x;
    if (idx >= C_DIM * C_DIM) return;
    int n = idx / C_DIM, k = idx % C_DIM;
    WT[(size_t)n * C_DIM + k] = (_Float16)W[(size_t)k * C_DIM + n];
}

// ---------------------------------------------------------------------------
// Depthwise 3x3 conv + BN, emit f16 token matrix with cls token and zero pad
// out[(b*tokPad + tok)*C + c]
// ---------------------------------------------------------------------------
__global__ void dwconv_bn_kernel(const float* __restrict__ x,
                                 const float* __restrict__ w,
                                 const float* __restrict__ gamma,
                                 const float* __restrict__ beta,
                                 const float* __restrict__ mean,
                                 const float* __restrict__ var,
                                 _Float16* __restrict__ out,
                                 int stride, int outHW, int tokPad) {
    int idx = blockIdx.x * blockDim.x + threadIdx.x;
    int total = NB * tokPad * C_DIM;
    if (idx >= total) return;
    int c   = idx % C_DIM;
    int tok = (idx / C_DIM) % tokPad;
    int b   = idx / (C_DIM * tokPad);
    int realTok = 1 + outHW * outHW;
    if (tok >= realTok) { out[idx] = (_Float16)0.f; return; }
    if (tok == 0) {
        out[idx] = (_Float16)x[((size_t)b * QTOK) * C_DIM + c];
        return;
    }
    int t = tok - 1;
    int ro = t / outHW, co = t % outHW;
    int ri0 = ro * stride - 1, ci0 = co * stride - 1;
    float acc = 0.f;
#pragma unroll
    for (int dr = 0; dr < 3; ++dr) {
        int ri = ri0 + dr;
        if (ri < 0 || ri >= 32) continue;
#pragma unroll
        for (int dc = 0; dc < 3; ++dc) {
            int ci = ci0 + dc;
            if (ci < 0 || ci >= 32) continue;
            float xv = x[((size_t)b * QTOK + 1 + ri * 32 + ci) * C_DIM + c];
            acc += xv * w[(dr * 3 + dc) * C_DIM + c];
        }
    }
    float inv = gamma[c] * rsqrtf(var[c] + 1e-5f);
    out[idx] = (_Float16)(acc * inv + (beta[c] - mean[c] * inv));
}

// ---------------------------------------------------------------------------
// A-fragment gather: 16x32 f16 tile in contiguous k order (B loads use the same order).
// lane m=lane&15 is the row; j0..15 -> k = hi*16+j
// ---------------------------------------------------------------------------
__device__ inline v16h load_a_frag(const _Float16* __restrict__ A, int lane, int k0) {
    int m = lane & 15, hi = lane >> 4;
    const _Float16* p = A + (size_t)m * C_DIM + k0 + hi * 16;
    v8h lo  = *(const v8h*)p;
    v8h hi8 = *(const v8h*)(p + 8);
    v16h r;
#pragma unroll
    for (int j = 0; j < 8; ++j) { r[j] = lo[j]; r[j + 8] = hi8[j]; }
    return r;
}

// ---------------------------------------------------------------------------
// GEMM: Out[M x 384] = A[M x 384] @ W  (WT is n-major f16)
// 6 waves per WG; each wave owns a 32-row x 64-col tile (B fragments reused
// across two A fragments -> 8 wmma per 12 b128 loads per k-step).
// Optionally store transposed per-batch ([ (b*384+n)*tokPerB + tok ]) for V.
// ---------------------------------------------------------------------------
__global__ __launch_bounds__(192) void gemm_wmma_f16(
        const _Float16* __restrict__ A, const _Float16* __restrict__ WT,
        _Float16* __restrict__ Out, int transOut, int tokPerB) {
    int lane = threadIdx.x & 31;
    int wave = threadIdx.x >> 5;       // 0..5
    int row0 = blockIdx.x * 32;
    int n0 = wave * 64;
    int m = lane & 15, hi = lane >> 4;
    const _Float16* Arow0 = A + (size_t)row0 * C_DIM;
    const _Float16* Arow1 = Arow0 + (size_t)16 * C_DIM;
    v8f acc0[4] = {};
    v8f acc1[4] = {};
    for (int k0 = 0; k0 < C_DIM; k0 += 32) {
        v16h a0 = load_a_frag(Arow0, lane, k0);
        v16h a1 = load_a_frag(Arow1, lane, k0);
#pragma unroll
        for (int t = 0; t < 4; ++t) {
            const _Float16* bp = WT + (size_t)(n0 + t * 16 + m) * C_DIM + k0 + hi * 16;
            v16h bfrag = *(const v16h*)bp;
            acc0[t] = __builtin_amdgcn_wmma_f32_16x16x32_f16(
                false, a0, false, bfrag, (short)0, acc0[t], false, false);
            acc1[t] = __builtin_amdgcn_wmma_f32_16x16x32_f16(
                false, a1, false, bfrag, (short)0, acc1[t], false, false);
        }
    }
#pragma unroll
    for (int half = 0; half < 2; ++half) {
#pragma unroll
        for (int t = 0; t < 4; ++t) {
            v8f acc = half ? acc1[t] : acc0[t];
            int gn = n0 + t * 16 + m;
#pragma unroll
            for (int r = 0; r < 8; ++r) {
                int grow = row0 + half * 16 + r + 8 * hi;
                _Float16 hv = (_Float16)acc[r];
                if (!transOut) {
                    Out[(size_t)grow * C_DIM + gn] = hv;
                } else {
                    int bb = grow / tokPerB;
                    int tk = grow - bb * tokPerB;
                    Out[((size_t)bb * C_DIM + gn) * tokPerB + tk] = hv;
                }
            }
        }
    }
}

// Final GEMM: f32 output + bias, guard padded rows, write to d_out
__global__ __launch_bounds__(192) void gemm_wmma_out(
        const _Float16* __restrict__ A, const _Float16* __restrict__ WT,
        const float* __restrict__ bias, float* __restrict__ Out) {
    int lane = threadIdx.x & 31;
    int wave = threadIdx.x >> 5;
    int row0 = blockIdx.x * 32;
    int n0 = wave * 64;
    int m = lane & 15, hi = lane >> 4;
    const _Float16* Arow0 = A + (size_t)row0 * C_DIM;
    const _Float16* Arow1 = Arow0 + (size_t)16 * C_DIM;
    v8f acc0[4] = {};
    v8f acc1[4] = {};
    for (int k0 = 0; k0 < C_DIM; k0 += 32) {
        v16h a0 = load_a_frag(Arow0, lane, k0);
        v16h a1 = load_a_frag(Arow1, lane, k0);
#pragma unroll
        for (int t = 0; t < 4; ++t) {
            const _Float16* bp = WT + (size_t)(n0 + t * 16 + m) * C_DIM + k0 + hi * 16;
            v16h bfrag = *(const v16h*)bp;
            acc0[t] = __builtin_amdgcn_wmma_f32_16x16x32_f16(
                false, a0, false, bfrag, (short)0, acc0[t], false, false);
            acc1[t] = __builtin_amdgcn_wmma_f32_16x16x32_f16(
                false, a1, false, bfrag, (short)0, acc1[t], false, false);
        }
    }
#pragma unroll
    for (int half = 0; half < 2; ++half) {
#pragma unroll
        for (int t = 0; t < 4; ++t) {
            v8f acc = half ? acc1[t] : acc0[t];
            int gn = n0 + t * 16 + m;
            float bv = bias[gn];
#pragma unroll
            for (int r = 0; r < 8; ++r) {
                int grow = row0 + half * 16 + r + 8 * hi;
                int bb = grow / QPAD;
                int tk = grow - bb * QPAD;
                if (tk < QTOK)
                    Out[((size_t)bb * QTOK + tk) * C_DIM + gn] = acc[r] + bv;
            }
        }
    }
}

// ---------------------------------------------------------------------------
// Flash attention: 1 wave per (qtile, head, batch). 9 chunks of 32 keys.
// qp: [b,QPAD,384] f16 ; kp: [b,KPAD,384] f16 ; vpT: [b,384,KPAD] f16
// ao: [b,QPAD,384] f16
// ---------------------------------------------------------------------------
__global__ __launch_bounds__(32) void attn_wmma(
        const _Float16* __restrict__ qp, const _Float16* __restrict__ kp,
        const _Float16* __restrict__ vpT, _Float16* __restrict__ ao) {
    __shared__ __align__(32) float    Sbuf[16 * 32];
    __shared__ __align__(32) _Float16 Pbuf[16 * 32];
    __shared__ float rowCorr[16];
    __shared__ float rowL[16];

    int lane = threadIdx.x & 31;
    int m = lane & 15, hi = lane >> 4;
    int mt = blockIdx.x, h = blockIdx.y, b = blockIdx.z;

    const _Float16* Qbase = qp + ((size_t)b * QPAD + mt * 16) * C_DIM + h * DH;
    v16h qa0 = load_a_frag(Qbase, lane, 0);
    v16h qa1 = load_a_frag(Qbase, lane, 32);

    const _Float16* Kb = kp + ((size_t)b * KPAD) * C_DIM + h * DH;
    const _Float16* Vb = vpT + ((size_t)b * C_DIM + h * DH) * KPAD;

    v8f o[4] = {};
    float mrun = -1e30f, lrun = 0.f;

    for (int kc = 0; kc < KPAD / 32; ++kc) {
        int t0 = kc * 32;
        if (kc + 1 < KPAD / 32) {
            __builtin_prefetch(Kb + (size_t)(t0 + 32) * C_DIM, 0, 1);
            __builtin_prefetch(Vb + (size_t)m * KPAD + t0 + 32, 0, 1);
        }

        // ---- S = scale * Q @ K^T  (16 x 32 tile) ----
        v8f s0 = {}, s1 = {};
#pragma unroll
        for (int nt = 0; nt < 2; ++nt) {
            const _Float16* kbp = Kb + (size_t)(t0 + nt * 16 + m) * C_DIM;
            v16h b0 = *(const v16h*)(kbp + hi * 16);        // dh 0..31
            v16h b1 = *(const v16h*)(kbp + 32 + hi * 16);   // dh 32..63
            v8f& ss = nt ? s1 : s0;
            ss = __builtin_amdgcn_wmma_f32_16x16x32_f16(false, qa0, false, b0, (short)0, ss, false, false);
            ss = __builtin_amdgcn_wmma_f32_16x16x32_f16(false, qa1, false, b1, (short)0, ss, false, false);
        }
#pragma unroll
        for (int r = 0; r < 8; ++r) {
            Sbuf[(r + 8 * hi) * 32 + m]      = s0[r] * ATT_SCALE;
            Sbuf[(r + 8 * hi) * 32 + 16 + m] = s1[r] * ATT_SCALE;
        }
        __syncthreads();

        // ---- online softmax: lanes 0..15 each own one row ----
        if (lane < 16) {
            float sv[32];
            float cmax = -1e30f;
#pragma unroll
            for (int j = 0; j < 32; ++j) {
                float xv = Sbuf[lane * 32 + j];
                if (t0 + j >= KTOK) xv = -1e30f;   // mask padded keys
                sv[j] = xv;
                cmax = fmaxf(cmax, xv);
            }
            float mnew = fmaxf(mrun, cmax);
            float corr = __expf(mrun - mnew);
            float lsum = 0.f;
#pragma unroll
            for (int j = 0; j < 32; ++j) {
                float p = __expf(sv[j] - mnew);
                lsum += p;
                Pbuf[lane * 32 + j] = (_Float16)p;
            }
            lrun = lrun * corr + lsum;
            mrun = mnew;
            rowCorr[lane] = corr;
        }
        __syncthreads();

        // ---- rescale O and accumulate O += P @ V ----
#pragma unroll
        for (int r = 0; r < 8; ++r) {
            float cf = rowCorr[r + 8 * hi];
            o[0][r] *= cf; o[1][r] *= cf; o[2][r] *= cf; o[3][r] *= cf;
        }
        v16h pa;
        {
            v8h lo  = *(const v8h*)&Pbuf[m * 32 + hi * 16];
            v8h hi8 = *(const v8h*)&Pbuf[m * 32 + hi * 16 + 8];
#pragma unroll
            for (int j = 0; j < 8; ++j) { pa[j] = lo[j]; pa[j + 8] = hi8[j]; }
        }
#pragma unroll
        for (int nt = 0; nt < 4; ++nt) {
            v16h vfrag = *(const v16h*)(Vb + (size_t)(nt * 16 + m) * KPAD + t0 + hi * 16);
            o[nt] = __builtin_amdgcn_wmma_f32_16x16x32_f16(
                false, pa, false, vfrag, (short)0, o[nt], false, false);
        }
        __syncthreads();
    }

    if (lane < 16) rowL[lane] = lrun;
    __syncthreads();
#pragma unroll
    for (int nt = 0; nt < 4; ++nt) {
        int gn = h * DH + nt * 16 + m;
#pragma unroll
        for (int r = 0; r < 8; ++r) {
            int row = r + 8 * hi;
            int tok = mt * 16 + row;
            float val = o[nt][r] * (1.f / rowL[row]);
            if (tok >= QTOK) val = 0.f;   // sanitize padded rows
            ao[((size_t)b * QPAD + tok) * C_DIM + gn] = (_Float16)val;
        }
    }
}

// ---------------------------------------------------------------------------
extern "C" void kernel_launch(void* const* d_in, const int* in_sizes, int n_in,
                              void* d_out, int out_size, void* d_ws, size_t ws_size,
                              hipStream_t stream) {
    (void)in_sizes; (void)n_in; (void)out_size; (void)ws_size;
    const float* x   = (const float*)d_in[0];
    const float* cwq = (const float*)d_in[1];
    const float* gq  = (const float*)d_in[2];
    const float* bq  = (const float*)d_in[3];
    const float* mq  = (const float*)d_in[4];
    const float* vq  = (const float*)d_in[5];
    const float* Wq  = (const float*)d_in[6];
    const float* cwk = (const float*)d_in[7];
    const float* gk  = (const float*)d_in[8];
    const float* bk  = (const float*)d_in[9];
    const float* mk  = (const float*)d_in[10];
    const float* vk  = (const float*)d_in[11];
    const float* Wk  = (const float*)d_in[12];
    const float* cwv = (const float*)d_in[13];
    const float* gv  = (const float*)d_in[14];
    const float* bv  = (const float*)d_in[15];
    const float* mv  = (const float*)d_in[16];
    const float* vv  = (const float*)d_in[17];
    const float* Wv  = (const float*)d_in[18];
    const float* Wp  = (const float*)d_in[19];
    const float* bp  = (const float*)d_in[20];

    char* ws = (char*)d_ws;
    size_t off = 0;
    auto alloc = [&](size_t bytes) -> void* {
        void* p = ws + off;
        off = (off + bytes + 255) & ~(size_t)255;
        return p;
    };
    const size_t wbytes = (size_t)C_DIM * C_DIM * sizeof(_Float16);
    _Float16* WqT = (_Float16*)alloc(wbytes);
    _Float16* WkT = (_Float16*)alloc(wbytes);
    _Float16* WvT = (_Float16*)alloc(wbytes);
    _Float16* WpT = (_Float16*)alloc(wbytes);
    _Float16* qh  = (_Float16*)alloc((size_t)NB * QPAD * C_DIM * sizeof(_Float16));
    _Float16* kh  = (_Float16*)alloc((size_t)NB * KPAD * C_DIM * sizeof(_Float16));
    _Float16* vh  = (_Float16*)alloc((size_t)NB * KPAD * C_DIM * sizeof(_Float16));
    _Float16* qpP = (_Float16*)alloc((size_t)NB * QPAD * C_DIM * sizeof(_Float16));
    _Float16* kpP = (_Float16*)alloc((size_t)NB * KPAD * C_DIM * sizeof(_Float16));
    _Float16* vpT = (_Float16*)alloc((size_t)NB * C_DIM * KPAD * sizeof(_Float16));
    _Float16* ao  = (_Float16*)alloc((size_t)NB * QPAD * C_DIM * sizeof(_Float16));

    int wthreads = C_DIM * C_DIM;
    pack_wt_kernel<<<(wthreads + 255) / 256, 256, 0, stream>>>(Wq, WqT);
    pack_wt_kernel<<<(wthreads + 255) / 256, 256, 0, stream>>>(Wk, WkT);
    pack_wt_kernel<<<(wthreads + 255) / 256, 256, 0, stream>>>(Wv, WvT);
    pack_wt_kernel<<<(wthreads + 255) / 256, 256, 0, stream>>>(Wp, WpT);

    int totQ = NB * QPAD * C_DIM;
    int totK = NB * KPAD * C_DIM;
    dwconv_bn_kernel<<<(totQ + 255) / 256, 256, 0, stream>>>(x, cwq, gq, bq, mq, vq, qh, 1, 32, QPAD);
    dwconv_bn_kernel<<<(totK + 255) / 256, 256, 0, stream>>>(x, cwk, gk, bk, mk, vk, kh, 2, 16, KPAD);
    dwconv_bn_kernel<<<(totK + 255) / 256, 256, 0, stream>>>(x, cwv, gv, bv, mv, vv, vh, 2, 16, KPAD);

    gemm_wmma_f16<<<NB * QPAD / 32, 192, 0, stream>>>(qh, WqT, qpP, 0, 1);
    gemm_wmma_f16<<<NB * KPAD / 32, 192, 0, stream>>>(kh, WkT, kpP, 0, 1);
    gemm_wmma_f16<<<NB * KPAD / 32, 192, 0, stream>>>(vh, WvT, vpT, 1, KPAD);

    dim3 ag(QPAD / 16, N_HEADS, NB);
    attn_wmma<<<ag, 32, 0, stream>>>(qpP, kpP, vpT, ao);

    gemm_wmma_out<<<NB * QPAD / 32, 192, 0, stream>>>(ao, WpT, bp, (float*)d_out);
}
